// StackMSA_86758339379251
// MI455X (gfx1250) — matrix operand order repaired and checked by reading, not compiled, not run on hardware
//
#include <hip/hip_runtime.h>

typedef unsigned short b16;
typedef __attribute__((ext_vector_type(16))) __bf16 v16bf;
typedef __attribute__((ext_vector_type(16))) unsigned short v16u;
typedef __attribute__((ext_vector_type(8))) float v8f;

__device__ __forceinline__ b16 f2bf(float f) {
  unsigned int u = __float_as_uint(f);
  u += 0x7FFFu + ((u >> 16) & 1u);   // round-to-nearest-even
  return (b16)(u >> 16);
}
__device__ __forceinline__ float bf2f(b16 h) {
  return __uint_as_float(((unsigned int)h) << 16);
}

// ---- WMMA operand builders, per gfx1250 VGPR layouts (CDNA5 ISA) ----
// A (16-bit, 16x32): lane m=lane&15, g=lane>>4; element e -> k = (e<8?0:16)+g*8+(e&7)
// (two contiguous 8-half runs per lane -> vectorizes to 2x ds_load_b128)
__device__ __forceinline__ v16bf make_abf(const b16* base, int stride) {
  int lane = threadIdx.x & 31;
  int m = lane & 15, g = lane >> 4;
  v16u u;
#pragma unroll
  for (int e = 0; e < 16; ++e) {
    int k = ((e & 8) << 1) + g * 8 + (e & 7);
    u[e] = base[m * stride + k];
  }
  return __builtin_bit_cast(v16bf, u);
}
// A operand from f32 LDS with per-lane scale (softmax 1/sum folded in)
__device__ __forceinline__ v16bf make_abf_f32s(const float* base, int stride, float scale) {
  int lane = threadIdx.x & 31;
  int m = lane & 15, g = lane >> 4;
  v16u u;
#pragma unroll
  for (int e = 0; e < 16; ++e) {
    int k = ((e & 8) << 1) + g * 8 + (e & 7);
    u[e] = f2bf(base[m * stride + k] * scale);
  }
  return __builtin_bit_cast(v16bf, u);
}
// B (16-bit, 32x16) from K-contiguous (transposed) storage:
// B[k][n] = base[n*stride + k], k = (e<8?0:16)+g*8+(e&7)  (two 8-half runs -> 2x ds_load_b128)
__device__ __forceinline__ v16bf make_bbf_T(const b16* base, int stride) {
  int lane = threadIdx.x & 31;
  int n = lane & 15, g = lane >> 4;
  v16u u;
#pragma unroll
  for (int e = 0; e < 16; ++e) u[e] = base[n * stride + ((e & 8) << 1) + g * 8 + (e & 7)];
  return __builtin_bit_cast(v16bf, u);
}
#define WMMA_BF16(a, b, c) \
  __builtin_amdgcn_wmma_f32_16x16x32_bf16(false, (a), false, (b), (short)0, (c), false, false)

// ======================= generic bf16 WMMA GEMM =======================
// C[M,N] = A[M,K] @ W[K,N] + bias, with W supplied TRANSPOSED as BT[N][K] (ldb = K stride).
// Optional exact-GELU ; optional f32 residual +=.
// Block tile 128x64, 8 waves (4x2), wave tile 32x32, K staged in 256-wide LDS panels.
// Requires M%128==0, N%64==0, K%256==0. Dynamic LDS = 96KB.
template <bool GELU, bool RESID>
__global__ __launch_bounds__(256) void gemm_bf16_kernel(
    const b16* __restrict__ A, int lda, const b16* __restrict__ BT, int ldb,
    const float* __restrict__ bias, b16* __restrict__ Cb, float* __restrict__ Cf,
    int ldc, int K) {
  extern __shared__ char gsm[];
  b16* As = (b16*)gsm;             // [128][256]
  b16* Bs = (b16*)(gsm + 65536);   // [64][256]  (K-contiguous per C column)
  const int row0 = blockIdx.y * 128;
  const int col0 = blockIdx.x * 64;
  const int tid = threadIdx.x;
  const int wid = tid >> 5, lane = tid & 31;
  const int wm = wid & 3, wn = wid >> 2;
  const int m = lane & 15, g = lane >> 4;
  v8f acc[2][2];
#pragma unroll
  for (int i = 0; i < 2; ++i)
#pragma unroll
    for (int j = 0; j < 2; ++j) acc[i][j] = (v8f){};

  for (int kc = 0; kc < K; kc += 256) {
#pragma unroll
    for (int i = 0; i < 16; ++i) {  // A panel: 128x256 = 4096 x 16B chunks
      int t = tid + i * 256;
      int r = t >> 5, cc = (t & 31) * 8;
      *(uint4*)&As[r * 256 + cc] =
          *(const uint4*)&A[(size_t)(row0 + r) * lda + kc + cc];
    }
#pragma unroll
    for (int i = 0; i < 8; ++i) {  // B panel: 64x256 = 2048 x 16B chunks
      int t = tid + i * 256;
      int n = t >> 5, cc = (t & 31) * 8;
      *(uint4*)&Bs[n * 256 + cc] =
          *(const uint4*)&BT[(size_t)(col0 + n) * ldb + kc + cc];
    }
    if (kc + 256 < K)  // prefetch next K-chunk of A (global_prefetch_b8)
      __builtin_prefetch(&A[(size_t)(row0 + (tid >> 1)) * lda + kc + 256 + (tid & 1) * 128], 0, 1);
    __syncthreads();
#pragma unroll
    for (int ks = 0; ks < 256; ks += 32) {
      v16bf a0 = make_abf(As + (wm * 32) * 256 + ks, 256);
      v16bf a1 = make_abf(As + (wm * 32 + 16) * 256 + ks, 256);
      v16bf b0 = make_bbf_T(Bs + (wn * 32) * 256 + ks, 256);
      v16bf b1 = make_bbf_T(Bs + (wn * 32 + 16) * 256 + ks, 256);
      acc[0][0] = WMMA_BF16(a0, b0, acc[0][0]);
      acc[0][1] = WMMA_BF16(a0, b1, acc[0][1]);
      acc[1][0] = WMMA_BF16(a1, b0, acc[1][0]);
      acc[1][1] = WMMA_BF16(a1, b1, acc[1][1]);
    }
    __syncthreads();
  }
#pragma unroll
  for (int mt = 0; mt < 2; ++mt)
#pragma unroll
    for (int nt = 0; nt < 2; ++nt)
#pragma unroll
      for (int v = 0; v < 8; ++v) {
        int r = row0 + wm * 32 + mt * 16 + v + 8 * g;
        int c = col0 + wn * 32 + nt * 16 + m;
        float x = acc[mt][nt][v] + bias[c];
        if (GELU) x = 0.5f * x * (1.0f + erff(x * 0.70710678118f));
        if (RESID)
          Cf[(size_t)r * ldc + c] += x;
        else
          Cb[(size_t)r * ldc + c] = f2bf(x);
      }
}

// ======================= fused windowed attention =======================
// one workgroup per (window, head); 8 waves, each wave does 2 query tiles.
// LDS: Q[256][32], K[256][32], V^T[32][256] (bf16), S[8 waves][16][256] (f32).
__global__ __launch_bounds__(256) void win_attn_kernel(
    const b16* __restrict__ qkv /*[win*256][768]*/, b16* __restrict__ out /*[win*256][256]*/) {
  extern __shared__ char smem[];
  b16* Qs = (b16*)smem;            // [256][32]
  b16* Ks = Qs + 256 * 32;         // [256][32]
  b16* Vt = Ks + 256 * 32;         // [32][256]  (transposed: Vt[d][tok])
  float* Sx = (float*)(smem + 3 * 256 * 32 * 2);  // 8 waves x 16x256 f32
  const int win = blockIdx.x, head = blockIdx.y;
  const int tid = threadIdx.x;
  const int wid = tid >> 5, lane = tid & 31;
  const int m = lane & 15, g = lane >> 4;
  const float qscale = 0.17677669529663687f;  // 1/sqrt(32)

  for (int i = tid; i < 256 * 32; i += 256) {
    int tok = i >> 5, d = i & 31;
    size_t base = ((size_t)win * 256 + tok) * 768 + head * 32 + d;
    Qs[i] = f2bf(bf2f(qkv[base]) * qscale);
    Ks[i] = qkv[base + 256];
    Vt[d * 256 + tok] = qkv[base + 512];
  }
  __syncthreads();

  float* Sw = Sx + wid * 16 * 256;
  for (int it = 0; it < 2; ++it) {
    const int mt = wid + it * 8;
    // S = (Q*scale) K^T
    v16bf a = make_abf(Qs + mt * 16 * 32, 32);
    for (int kt = 0; kt < 16; ++kt) {
      v16bf b = make_bbf_T(Ks + kt * 16 * 32, 32);
      v8f c = (v8f){};
      c = WMMA_BF16(a, b, c);
#pragma unroll
      for (int v = 0; v < 8; ++v) Sw[(v + 8 * g) * 256 + kt * 16 + m] = c[v];
    }
    __syncthreads();
    // softmax: lane handles row m, half g (128 cols as 32 float4); combine via shfl_xor(16).
    // exp() written unnormalized; 1/sum folded into the P->bf16 conversion below.
    float inv;
    {
      float4* row4 = (float4*)(Sw + m * 256 + g * 128);
      float mx = -1e30f;
#pragma unroll 8
      for (int j = 0; j < 32; ++j) {
        float4 v = row4[j];
        mx = fmaxf(mx, fmaxf(fmaxf(v.x, v.y), fmaxf(v.z, v.w)));
      }
      mx = fmaxf(mx, __shfl_xor(mx, 16, 32));
      float sum = 0.f;
#pragma unroll 8
      for (int j = 0; j < 32; ++j) {
        float4 v = row4[j];
        v.x = __expf(v.x - mx);
        v.y = __expf(v.y - mx);
        v.z = __expf(v.z - mx);
        v.w = __expf(v.w - mx);
        row4[j] = v;
        sum += (v.x + v.y) + (v.z + v.w);
      }
      sum += __shfl_xor(sum, 16, 32);
      inv = 1.0f / sum;
    }
    __syncthreads();
    // O = P V  (P scaled by inv during bf16 conversion; V^T gives contiguous B loads)
    v8f o0 = (v8f){}, o1 = (v8f){};
#pragma unroll
    for (int kc = 0; kc < 8; ++kc) {
      v16bf pa = make_abf_f32s(Sw + kc * 32, 256, inv);
      v16bf vb0 = make_bbf_T(Vt + kc * 32, 256);
      v16bf vb1 = make_bbf_T(Vt + 16 * 256 + kc * 32, 256);
      o0 = WMMA_BF16(pa, vb0, o0);
      o1 = WMMA_BF16(pa, vb1, o1);
    }
#pragma unroll
    for (int v = 0; v < 8; ++v) {
      size_t r = ((size_t)win * 256 + mt * 16 + v + 8 * g) * 256 + head * 32;
      out[r + m] = f2bf(o0[v]);
      out[r + 16 + m] = f2bf(o1[v]);
    }
    __syncthreads();
  }
}

// ======================= LayerNorm (MODE 1 = fine-region gather) =======================
template <int MODE>
__global__ __launch_bounds__(256) void ln_kernel(const float* __restrict__ x,
                                                 const float* __restrict__ gg,
                                                 const float* __restrict__ bb,
                                                 b16* __restrict__ out) {
  const int row = blockIdx.x, t = threadIdx.x;
  __shared__ float red[256];
  size_t src;
  if (MODE == 0) {
    src = (size_t)row * 256;
  } else {  // fine: row = group*16 + j ; group=(b,h,w,r1,r2), j=(p1,p2)
    int grp = row >> 4, j = row & 15;
    int r2 = grp & 3, tmp = grp >> 2;
    int r1 = tmp & 3; tmp >>= 2;
    int w = tmp & 7; tmp >>= 3;
    int h = tmp & 7; int b = tmp >> 3;
    int p1 = j >> 2, p2 = j & 3;
    int win = (b * 8 + h) * 8 + w;
    int tok = (r1 * 4 + p1) * 16 + (r2 * 4 + p2);
    src = ((size_t)win * 256 + tok) * 256;
  }
  float v = x[src + t];
  red[t] = v;
  __syncthreads();
  for (int s = 128; s > 0; s >>= 1) {
    if (t < s) red[t] += red[t + s];
    __syncthreads();
  }
  float mean = red[0] * (1.0f / 256.0f);
  __syncthreads();
  float d = v - mean;
  red[t] = d * d;
  __syncthreads();
  for (int s = 128; s > 0; s >>= 1) {
    if (t < s) red[t] += red[t + s];
    __syncthreads();
  }
  float var = red[0] * (1.0f / 256.0f);
  out[(size_t)row * 256 + t] = f2bf(d * rsqrtf(var + 1e-5f) * gg[t] + bb[t]);
}

// ======================= small helper kernels =======================
__global__ void pe_kernel(const float* __restrict__ w1, const float* __restrict__ b1,
                          const float* __restrict__ w2, float* __restrict__ pe0,
                          float* __restrict__ pe1) {
  const int s = blockIdx.y, n = blockIdx.x, t = threadIdx.x;
  __shared__ float hid[512];
  float cy = ((float)(n >> 4) - 8.0f) * 0.125f;
  float cx = ((float)(n & 15) - 8.0f) * 0.125f;
  for (int j = t; j < 512; j += 256) {
    float hv = cy * w1[s * 1024 + j] + cx * w1[s * 1024 + 512 + j] + b1[s * 512 + j];
    hid[j] = fmaxf(hv, 0.0f);
  }
  __syncthreads();
  float acc = 0.0f;
  const float* w2s = w2 + (size_t)s * 512 * 256;
  for (int j = 0; j < 512; ++j) acc += hid[j] * w2s[j * 256 + t];
  (s == 0 ? pe0 : pe1)[n * 256 + t] = acc;
}

// f32 W[K][N] -> bf16 W^T[N][K]
__global__ void cvt_t_kernel(const float* __restrict__ src, b16* __restrict__ dst,
                             int K, int N) {
  int i = blockIdx.x * 256 + threadIdx.x;
  if (i < K * N) {
    int k = i / N, n = i - k * N;
    dst[(size_t)n * K + k] = f2bf(src[i]);
  }
}

__global__ void add_pe0_kernel(const float* __restrict__ s0, const float* __restrict__ pe0,
                               float* __restrict__ x0) {
  size_t i = (size_t)blockIdx.x * 256 + threadIdx.x;
  x0[i] = s0[i] + pe0[i & 65535];
}

__global__ void pool_kernel(const float* __restrict__ s1, const float* __restrict__ pe1,
                            const float* __restrict__ x0, float* __restrict__ x1) {
  const int idx = blockIdx.x, c = threadIdx.x;  // idx = win1*256+tok
  const int win1 = idx >> 8, tok = idx & 255;
  const int my = tok >> 4, mx = tok & 15;
  const int b = win1 >> 2, wrem = win1 & 3, wy = wrem >> 1, wx = wrem & 1;
  const int gy = wy * 16 + my, gx = wx * 16 + mx;
  const int h = gy >> 2, r1 = gy & 3, w = gx >> 2, r2 = gx & 3;
  const int win0 = (b * 8 + h) * 8 + w;
  float mv = -1e30f;
#pragma unroll
  for (int p1 = 0; p1 < 4; ++p1)
#pragma unroll
    for (int p2 = 0; p2 < 4; ++p2) {
      int t0 = (r1 * 4 + p1) * 16 + (r2 * 4 + p2);
      mv = fmaxf(mv, x0[((size_t)win0 * 256 + t0) * 256 + c]);
    }
  x1[(size_t)idx * 256 + c] = s1[(size_t)idx * 256 + c] + pe1[tok * 256 + c] + mv;
}

__global__ void copy_kernel(const float* __restrict__ src, float* __restrict__ dst) {
  size_t i = (size_t)blockIdx.x * 256 + threadIdx.x;
  dst[i] = src[i];
}

__global__ void gather_g_kernel(const float* __restrict__ x1, float* __restrict__ gbuf) {
  const int grp = blockIdx.x, c = threadIdx.x;
  int r2 = grp & 3, t1 = grp >> 2;
  int r1 = t1 & 3; t1 >>= 2;
  int w = t1 & 7; t1 >>= 3;
  int h = t1 & 7; int b = t1 >> 3;
  int gy = h * 4 + r1, gx = w * 4 + r2;
  int wy = gy >> 4, my = gy & 15, wx = gx >> 4, mx = gx & 15;
  int win = b * 4 + wy * 2 + wx, tok = my * 16 + mx;
  gbuf[(size_t)grp * 256 + c] = x1[((size_t)win * 256 + tok) * 256 + c];
}

__global__ void scatter_g_kernel(const float* __restrict__ gbuf, float* __restrict__ outp) {
  const int idx = blockIdx.x, c = threadIdx.x;  // idx = win1*256+tok
  const int win1 = idx >> 8, tok = idx & 255;
  const int my = tok >> 4, mx = tok & 15;
  const int b = win1 >> 2, wrem = win1 & 3, wy = wrem >> 1, wx = wrem & 1;
  const int gy = wy * 16 + my, gx = wx * 16 + mx;
  const int h = gy >> 2, r1 = gy & 3, w = gx >> 2, r2 = gx & 3;
  const int grp = (((b * 8 + h) * 8 + w) * 4 + r1) * 4 + r2;
  outp[(size_t)idx * 256 + c] = gbuf[(size_t)grp * 256 + c];
}

__global__ __launch_bounds__(128) void cross_attn_kernel(const b16* __restrict__ Qg,
                                                         const b16* __restrict__ KVf,
                                                         b16* __restrict__ out) {
  const int grp = blockIdx.x, t = threadIdx.x;  // 128 threads
  __shared__ float qv[256], sc[128], pr[128];
  qv[t] = bf2f(Qg[(size_t)grp * 256 + t]);
  qv[t + 128] = bf2f(Qg[(size_t)grp * 256 + t + 128]);
  __syncthreads();
  {
    int h = t >> 4, j = t & 15;
    const b16* krow = KVf + ((size_t)grp * 16 + j) * 512 + h * 32;
    float s = 0.0f;
    for (int d = 0; d < 32; ++d) s += qv[h * 32 + d] * bf2f(krow[d]);
    sc[t] = s * 0.17677669529663687f;
  }
  __syncthreads();
  if (t < 8) {
    float mx = -1e30f;
    for (int j = 0; j < 16; ++j) mx = fmaxf(mx, sc[t * 16 + j]);
    float sum = 0.0f;
    for (int j = 0; j < 16; ++j) {
      float e = __expf(sc[t * 16 + j] - mx);
      pr[t * 16 + j] = e;
      sum += e;
    }
    float inv = 1.0f / sum;
    for (int j = 0; j < 16; ++j) pr[t * 16 + j] *= inv;
  }
  __syncthreads();
  for (int c = t; c < 256; c += 128) {
    int hh = c >> 5;
    float o = 0.0f;
    for (int j = 0; j < 16; ++j)
      o += pr[hh * 16 + j] * bf2f(KVf[((size_t)grp * 16 + j) * 512 + 256 + c]);
    out[(size_t)grp * 256 + c] = f2bf(o);
  }
}

// ======================= host orchestration =======================
static void run_sattn(int s, float* X, int bn, b16* HB, b16* QKVB, b16* ATTB, b16* MID,
                      const float* ln1_g, const float* ln1_b, const b16* WQKVt,
                      const float* bqkv, const b16* WOt, const float* bo,
                      const float* ln2_g, const float* ln2_b, const b16* W1t,
                      const float* mb1, const b16* W2t, const float* mb2,
                      hipStream_t stream) {
  const int rows = bn * 256;
  ln_kernel<0><<<rows, 256, 0, stream>>>(X, ln1_g + s * 256, ln1_b + s * 256, HB);
  gemm_bf16_kernel<false, false><<<dim3(12, rows / 128), 256, 98304, stream>>>(
      HB, 256, WQKVt + s * 196608, 256, bqkv + s * 768, QKVB, nullptr, 768, 256);
  win_attn_kernel<<<dim3(bn, 8), 256, 180224, stream>>>(QKVB, ATTB);
  gemm_bf16_kernel<false, true><<<dim3(4, rows / 128), 256, 98304, stream>>>(
      ATTB, 256, WOt + s * 65536, 256, bo + s * 256, nullptr, X, 256, 256);
  ln_kernel<0><<<rows, 256, 0, stream>>>(X, ln2_g + s * 256, ln2_b + s * 256, HB);
  gemm_bf16_kernel<true, false><<<dim3(16, rows / 128), 256, 98304, stream>>>(
      HB, 256, W1t + s * 262144, 256, mb1 + s * 1024, MID, nullptr, 1024, 256);
  gemm_bf16_kernel<false, true><<<dim3(4, rows / 128), 256, 98304, stream>>>(
      MID, 1024, W2t + s * 262144, 1024, mb2 + s * 256, nullptr, X, 256, 1024);
}

extern "C" void kernel_launch(void* const* d_in, const int* in_sizes, int n_in,
                              void* d_out, int out_size, void* d_ws, size_t ws_size,
                              hipStream_t stream) {
  const float* scale0 = (const float*)d_in[0];
  const float* scale1 = (const float*)d_in[1];
  const float* pe_w1 = (const float*)d_in[2];
  const float* pe_b1 = (const float*)d_in[3];
  const float* pe_w2 = (const float*)d_in[4];
  const float* ln1_g = (const float*)d_in[5];
  const float* ln1_b = (const float*)d_in[6];
  const float* wqkv = (const float*)d_in[7];
  const float* bqkv = (const float*)d_in[8];
  const float* wo = (const float*)d_in[9];
  const float* bo = (const float*)d_in[10];
  const float* ln2_g = (const float*)d_in[11];
  const float* ln2_b = (const float*)d_in[12];
  const float* mw1 = (const float*)d_in[13];
  const float* mb1 = (const float*)d_in[14];
  const float* mw2 = (const float*)d_in[15];
  const float* mb2 = (const float*)d_in[16];
  (void)in_sizes; (void)n_in; (void)out_size;

  char* ws = (char*)d_ws;
  size_t off = 0;
  float* X0 = (float*)(ws + off); off += 67108864;    // [256win,256,256] f32
  float* X1 = (float*)(ws + off); off += 4194304;     // [16win,256,256] f32
  float* GBUF = (float*)(ws + off); off += 4194304;   // [4096,256] f32
  float* PE0 = (float*)(ws + off); off += 262144;
  float* PE1 = (float*)(ws + off); off += 262144;
  b16* WQKVt = (b16*)(ws + off); off += 786432;       // [2][768][256] bf16 (transposed)
  b16* WOt = (b16*)(ws + off); off += 262144;         // [2][256][256]
  b16* W1t = (b16*)(ws + off); off += 1048576;        // [2][1024][256]
  b16* W2t = (b16*)(ws + off); off += 1048576;        // [2][256][1024]
  b16* HB = (b16*)(ws + off); off += 35651584;        // LN out (FLN 32MB + GLN 2MB)
  char* POOL = ws + off; off += 134217728;            // phase-aliased scratch
  if (ws_size < off) return;

  // phase 1 aliases (self-attn blocks)
  b16* QKVB = (b16*)POOL;                 // [rows,768] bf16 (96MB @ scale0)
  b16* ATTB = (b16*)(POOL + 100663296);   // [rows,256] bf16 (32MB)
  b16* MID = (b16*)POOL;                  // [rows,1024] bf16 (128MB) -- after QKV/ATT dead
  // phase 2 aliases (cross-attention)
  b16* KVF = (b16*)POOL;                  // [65536,512] bf16 (64MB)
  b16* QG = (b16*)(POOL + 67108864);      // [4096,256] bf16
  b16* XO = (b16*)(POOL + 71303168);      // [4096,256] bf16
  b16* MIDc = (b16*)(POOL + 75497472);    // [4096,1024] bf16
  b16* FLN = HB;                          // [65536,256] bf16
  b16* GLN = (b16*)((char*)HB + 33554432);

  // position-embedding tables + transposed bf16 weight conversion
  pe_kernel<<<dim3(256, 2), 256, 0, stream>>>(pe_w1, pe_b1, pe_w2, PE0, PE1);
  for (int s = 0; s < 2; ++s) {
    cvt_t_kernel<<<768, 256, 0, stream>>>(wqkv + s * 196608, WQKVt + s * 196608, 256, 768);
    cvt_t_kernel<<<256, 256, 0, stream>>>(wo + s * 65536, WOt + s * 65536, 256, 256);
    cvt_t_kernel<<<1024, 256, 0, stream>>>(mw1 + s * 262144, W1t + s * 262144, 256, 1024);
    cvt_t_kernel<<<1024, 256, 0, stream>>>(mw2 + s * 262144, W2t + s * 262144, 1024, 256);
  }

  // x0 = scale0 + pe0 ; x1 = scale1 + pe1 + maxpool-propagate(x0)
  add_pe0_kernel<<<65536, 256, 0, stream>>>(scale0, PE0, X0);
  pool_kernel<<<4096, 256, 0, stream>>>(scale1, PE1, X0, X1);

  // self-attention blocks (in-place residual on X1 / X0)
  run_sattn(1, X1, 16, HB, QKVB, ATTB, MID, ln1_g, ln1_b, WQKVt, bqkv, WOt, bo,
            ln2_g, ln2_b, W1t, mb1, W2t, mb2, stream);
  run_sattn(0, X0, 256, HB, QKVB, ATTB, MID, ln1_g, ln1_b, WQKVt, bqkv, WOt, bo,
            ln2_g, ln2_b, W1t, mb1, W2t, mb2, stream);

  // o0 -> output (first 256 windows)
  copy_kernel<<<65536, 256, 0, stream>>>(X0, (float*)d_out);

  // one2one cross-attention: g queries its 4x4 fine region
  ln_kernel<1><<<65536, 256, 0, stream>>>(X0, ln1_g + 256, ln1_b + 256, FLN);
  gather_g_kernel<<<4096, 256, 0, stream>>>(X1, GBUF);
  ln_kernel<0><<<4096, 256, 0, stream>>>(GBUF, ln1_g + 256, ln1_b + 256, GLN);
  // K,V projection: rows 256..767 of transposed wqkv[1]
  gemm_bf16_kernel<false, false><<<dim3(8, 512), 256, 98304, stream>>>(
      FLN, 256, WQKVt + 196608 + 65536, 256, bqkv + 768 + 256, KVF, nullptr, 512, 256);
  gemm_bf16_kernel<false, false><<<dim3(4, 32), 256, 98304, stream>>>(
      GLN, 256, WQKVt + 196608, 256, bqkv + 768, QG, nullptr, 256, 256);
  cross_attn_kernel<<<4096, 128, 0, stream>>>(QG, KVF, XO);
  gemm_bf16_kernel<false, true><<<dim3(4, 32), 256, 98304, stream>>>(
      XO, 256, WOt + 65536, 256, bo + 256, nullptr, GBUF, 256, 256);
  ln_kernel<0><<<4096, 256, 0, stream>>>(GBUF, ln2_g + 256, ln2_b + 256, GLN);
  gemm_bf16_kernel<true, false><<<dim3(16, 32), 256, 98304, stream>>>(
      GLN, 256, W1t + 262144, 256, mb1 + 1024, MIDc, nullptr, 1024, 256);
  gemm_bf16_kernel<false, true><<<dim3(4, 32), 256, 98304, stream>>>(
      MIDc, 1024, W2t + 262144, 1024, mb2 + 256, nullptr, GBUF, 256, 1024);

  // g -> output (windows 256..271)
  scatter_g_kernel<<<4096, 256, 0, stream>>>(GBUF, (float*)d_out + 16777216);
}
